// ResonAItBrain_65549790871604
// MI455X (gfx1250) — matrix operand order repaired and checked by reading, not compiled, not run on hardware
//
#include <hip/hip_runtime.h>
#include <math.h>
#include <stdint.h>

typedef __attribute__((ext_vector_type(16))) _Float16 v16h;
typedef __attribute__((ext_vector_type(4)))  _Float16 v4h;
typedef __attribute__((ext_vector_type(8)))  float    v8f;
typedef __attribute__((ext_vector_type(4)))  float    v4f;

namespace {

constexpr int TB_M = 128, TB_N = 64, TB_K = 32;
constexpr int BN_  = 128;   // batch
constexpr int CN_  = 256;   // hidden channels
constexpr int LN_  = 512;   // sequence / freq dim
constexpr int MMAX = 64;    // max fourier modes
constexpr int FCHUNK = 16;  // batches per FFN/bind chunk

enum : int {
  GF_TRANS_A   = 1 << 0,
  GF_TRANS_C   = 1 << 1,
  GF_BIAS_ROW  = 1 << 2,
  GF_BIAS_COL  = 1 << 3,
  GF_GELU      = 1 << 4,
  GF_PRE_ADD   = 1 << 5,
  GF_POST_ADD  = 1 << 6,
  GF_ST_ADD    = 1 << 7,
  GF_ST_SUB    = 1 << 8,
  GF_B_CAT3    = 1 << 9,
  GF_B_HALF    = 1 << 10,  // B is _Float16, ldbc==1; async-to-LDS staging
  GF_B_STRIDED = 1 << 11,  // B column stride ldbc != 1 (mode-mix weights)
};

__device__ __forceinline__ float gelu_exact(float x) {
  return 0.5f * x * (1.0f + erff(x * 0.70710678118654752440f));
}

// low 32 bits of a generic pointer to __shared__ == LDS byte address (ISA 10.2)
__device__ __forceinline__ unsigned lds_addr32(const void* p) {
  return (unsigned)(uintptr_t)p;
}

// B-tile LDS indexing: K-major (async raw copy) vs N-major (v16h fragment reads)
template <bool KMAJOR>
__device__ __forceinline__ constexpr int boff(int kk, int nn) {
  return KMAJOR ? kk * (TB_N + 16) + nn : nn * (TB_K + 16) + kk;
}

// ---------------------------------------------------------------------------
// Generic f16-WMMA GEMM, fp32 in/out, fp32 accumulate; FLAGS compile-time.
//   C[z](M,N) = epilogue( A[z](M,K) * B[z](K,N) )
// All non-B_HALF call sites in this network are tile-aligned in M,N,K, so
// those instantiations carry only unpredicated vector loaders.
// ---------------------------------------------------------------------------
template <int FLAGS>
__global__ __launch_bounds__(256, 2)
void wmma_gemm_kernel(const float* __restrict__ Ap,
                      const float* __restrict__ Bp,
                      float* Cp,
                      const float* __restrict__ Bcat1,
                      const float* __restrict__ Bcat2,
                      const float* __restrict__ gates, int catRows, int gateStride,
                      int M, int N, int K,
                      int lda, int ldb, int ldbc, int ldc,
                      long long sA, long long sB, long long sC,
                      const float* __restrict__ bias,
                      const float* pre,  long long sPre,
                      const float* post, long long sPost)
{
  constexpr bool BH = (FLAGS & GF_B_HALF) != 0;

  __shared__ __align__(16) _Float16 sAt[TB_M][TB_K + 8];
  __shared__ __align__(32) _Float16 sBl[BH ? TB_K * (TB_N + 16)
                                           : TB_N * (TB_K + 16)];

  const int z    = blockIdx.z;
  const int m0   = blockIdx.y * TB_M;
  const int n0   = blockIdx.x * TB_N;
  const int tid  = threadIdx.x;
  const int lane = tid & 31;
  const int wave = tid >> 5;       // 0..7
  const int wrow = wave & 3;       // 4 wave-rows * 32 = 128
  const int wcol = wave >> 2;      // 2 wave-cols * 32 = 64
  const int g    = (lane >> 4) & 1;
  const int ml   = lane & 15;

  const float*    Az  = Ap + (long long)z * sA;
  const float*    Bz  = Bp + (long long)z * sB;
  const _Float16* BzH = (const _Float16*)Bz;                 // GF_B_HALF view
  const float*    B1z = (FLAGS & GF_B_CAT3) ? Bcat1 + (long long)z * sB : nullptr;
  const float*    B2z = (FLAGS & GF_B_CAT3) ? Bcat2 + (long long)z * sB : nullptr;

  const bool fullN = (n0 + TB_N) <= N;

  const v8f vzero = {0.f,0.f,0.f,0.f,0.f,0.f,0.f,0.f};
  v8f acc[2][2];
  acc[0][0] = vzero; acc[0][1] = vzero; acc[1][0] = vzero; acc[1][1] = vzero;

  for (int kb = 0; kb < K; kb += TB_K) {
    // ------------- stage A tile (TB_M x TB_K) as f16, vectorized -----------
    if constexpr (FLAGS & GF_TRANS_A) {
      #pragma unroll
      for (int it = 0; it < (TB_M * TB_K) / (256 * 4); ++it) {   // 4
        int idx = tid + it * 256;
        int mm4 = idx & 31;            // TB_M/4
        int kk  = idx >> 5;
        v4f v = *(const v4f*)(Az + (long long)(kb + kk) * lda + m0 + mm4 * 4);
        sAt[mm4 * 4 + 0][kk] = (_Float16)v.x;
        sAt[mm4 * 4 + 1][kk] = (_Float16)v.y;
        sAt[mm4 * 4 + 2][kk] = (_Float16)v.z;
        sAt[mm4 * 4 + 3][kk] = (_Float16)v.w;
      }
    } else {
      #pragma unroll
      for (int it = 0; it < (TB_M * TB_K) / (256 * 4); ++it) {   // 4
        int idx = tid + it * 256;
        int kk4 = idx & 7;             // TB_K/4
        int mm  = idx >> 3;
        v4f v = *(const v4f*)(Az + (long long)(m0 + mm) * lda + kb + kk4 * 4);
        v4h h;
        h.x = (_Float16)v.x; h.y = (_Float16)v.y;
        h.z = (_Float16)v.z; h.w = (_Float16)v.w;
        *(v4h*)&sAt[mm][kk4 * 4] = h;
      }
      if (kb + TB_K < K) {             // prefetch next A tile (global_prefetch)
        int kk4 = tid & 7, mm = tid >> 3;
        __builtin_prefetch(Az + (long long)(m0 + mm) * lda + (kb + TB_K) + kk4 * 4, 0, 0);
      }
    }

    // ------------- stage B tile (TB_K x TB_N) as f16 -----------------------
    if constexpr (BH) {
      if (fullN) {
        // CDNA5 async copy: 8 bytes/lane directly into LDS (ASYNCcnt-tracked)
        #pragma unroll
        for (int it = 0; it < (TB_K * TB_N) / (256 * 4); ++it) {   // 2
          int idx = tid + it * 256;
          int nn4 = idx & 15;          // TB_N/4
          int kk  = idx >> 4;
          unsigned srcOff = (unsigned)(((long long)(kb + kk) * ldb + n0 + nn4 * 4)
                                       * (long long)sizeof(_Float16));
          unsigned dstLds = lds_addr32(&sBl[boff<true>(kk, nn4 * 4)]);
          asm volatile("global_load_async_to_lds_b64 %0, %1, %2"
                       :: "v"(dstLds), "v"(srcOff), "s"(BzH)
                       : "memory");
        }
      } else {                         // partial N (fwd DFT with m=16)
        #pragma unroll
        for (int it = 0; it < (TB_K * TB_N) / 256; ++it) {
          int idx = tid + it * 256;
          int nn = idx % TB_N;
          int kk = idx / TB_N;
          int gn = n0 + nn;
          bool ok = gn < N;
          _Float16 v = BzH[(long long)(kb + kk) * ldb + (ok ? gn : 0)];
          sBl[boff<true>(kk, nn)] = ok ? v : (_Float16)0.0f;
        }
      }
    } else if constexpr (FLAGS & GF_B_CAT3) {
      #pragma unroll
      for (int it = 0; it < (TB_K * TB_N) / 256; ++it) {
        int idx = tid + it * 256;
        int nn = idx % TB_N;
        int kk = idx / TB_N;
        int gk = kb + kk;
        int seg = gk / catRows;
        int rr  = gk - seg * catRows;
        const float* base = (seg == 0) ? Bz : (seg == 1) ? B1z : B2z;
        float v = base[(long long)rr * ldb + n0 + nn] * gates[seg * gateStride + z];
        sBl[boff<false>(kk, nn)] = (_Float16)v;
      }
    } else if constexpr (FLAGS & GF_B_STRIDED) {   // mode-mix weights
      #pragma unroll
      for (int it = 0; it < (TB_K * TB_N) / 256; ++it) {
        int idx = tid + it * 256;
        int nn = idx % TB_N;
        int kk = idx / TB_N;
        float v = Bz[(long long)(kb + kk) * ldb + (long long)(n0 + nn) * ldbc];
        sBl[boff<false>(kk, nn)] = (_Float16)v;
      }
    } else {
      #pragma unroll
      for (int it = 0; it < (TB_K * TB_N) / (256 * 4); ++it) {   // 2
        int idx = tid + it * 256;
        int nn4 = idx & 15;            // TB_N/4
        int kk  = idx >> 4;
        v4f v = *(const v4f*)(Bz + (long long)(kb + kk) * ldb + n0 + nn4 * 4);
        sBl[boff<false>(kk, nn4 * 4 + 0)] = (_Float16)v.x;
        sBl[boff<false>(kk, nn4 * 4 + 1)] = (_Float16)v.y;
        sBl[boff<false>(kk, nn4 * 4 + 2)] = (_Float16)v.z;
        sBl[boff<false>(kk, nn4 * 4 + 3)] = (_Float16)v.w;
      }
    }

    if constexpr (BH) {
      if (fullN)
        asm volatile("s_wait_asynccnt 0" ::: "memory");
    }
    __syncthreads();

    // ---- gather fragments for gfx1250 WMMA 16x16x32 f16 (k = 16g + j in both A and B) ----
    v16h afr[2], bfr[2];
    #pragma unroll
    for (int t = 0; t < 2; ++t) {
      const int aRow = wrow * 32 + t * 16 + ml;
      #pragma unroll
      for (int j = 0; j < 16; ++j) {
        int kk = 16 * g + j;                    // A: lanes0-15 K{0..15}, 16-31 K{16..31}
        afr[t][j] = sAt[aRow][kk];
      }
      const int bCol = wcol * 32 + t * 16 + ml;
      if constexpr (BH) {
        #pragma unroll
        for (int j = 0; j < 16; ++j)
          bfr[t][j] = sBl[boff<true>(16 * g + j, bCol)];
      } else {
        // N-major layout: lane's 16 K-halves are one aligned contiguous v16h
        bfr[t] = *(const v16h*)&sBl[boff<false>(16 * g, bCol)];
      }
    }
    #pragma unroll
    for (int tm = 0; tm < 2; ++tm)
      #pragma unroll
      for (int tn = 0; tn < 2; ++tn)
        acc[tm][tn] = __builtin_amdgcn_wmma_f32_16x16x32_f16(
            false, afr[tm], false, bfr[tn], (short)0, acc[tm][tn], false, false);
    __syncthreads();
  }

  // ---- epilogue ----
  float* Cz = Cp + (long long)z * sC;
  const float* prez  = (FLAGS & GF_PRE_ADD)  ? pre  + (long long)z * sPre  : nullptr;
  const float* postz = (FLAGS & GF_POST_ADD) ? post + (long long)z * sPost : nullptr;

  #pragma unroll
  for (int tm = 0; tm < 2; ++tm) {
    #pragma unroll
    for (int tn = 0; tn < 2; ++tn) {
      #pragma unroll
      for (int r = 0; r < 8; ++r) {
        int row = m0 + wrow * 32 + tm * 16 + r + 8 * g;   // C/D: VGPR r -> M=r+8g
        int col = n0 + wcol * 32 + tn * 16 + ml;          //       lane  -> N
        if (row < M && col < N) {
          float v = acc[tm][tn][r];
          if constexpr (FLAGS & GF_BIAS_ROW) v += bias[row];
          if constexpr (FLAGS & GF_BIAS_COL) v += bias[col];
          if constexpr (FLAGS & GF_PRE_ADD)  v += prez[(long long)row * ldc + col];
          if constexpr (FLAGS & GF_GELU)     v = gelu_exact(v);
          if constexpr (FLAGS & GF_POST_ADD) v += postz[(long long)row * ldc + col];
          long long coff = (FLAGS & GF_TRANS_C)
                               ? (long long)col * ldc + row
                               : (long long)row * ldc + col;
          if constexpr (FLAGS & GF_ST_ADD)      Cz[coff] += v;
          else if constexpr (FLAGS & GF_ST_SUB) Cz[coff] -= v;
          else                                  Cz[coff]  = v;
        }
      }
    }
  }
}

// LayerNorm over the channel axis of (B, C, L): normalizes each (b, l) column.
__global__ void ln_channels_kernel(const float* __restrict__ in, float* __restrict__ out,
                                   const float* __restrict__ gam, const float* __restrict__ bet,
                                   int Bn, int Cn, int Ln)
{
  int idx = blockIdx.x * blockDim.x + threadIdx.x;
  if (idx >= Bn * Ln) return;
  int bb = idx / Ln, l = idx - bb * Ln;
  const float* p = in + (long long)bb * Cn * Ln + l;
  float s = 0.f, ss = 0.f;
  for (int c = 0; c < Cn; ++c) { float v = p[(long long)c * Ln]; s += v; ss += v * v; }
  float mu = s / Cn;
  float var = ss / Cn - mu * mu;
  float rstd = rsqrtf(var + 1e-5f);
  float* q = out + (long long)bb * Cn * Ln + l;
  for (int c = 0; c < Cn; ++c)
    q[(long long)c * Ln] = (p[(long long)c * Ln] - mu) * rstd * gam[c] + bet[c];
}

// gate[b] = sigmoid( mean_L(x[b]) . gw + gb )
__global__ void gate_kernel(const float* __restrict__ x, const float* __restrict__ gw,
                            const float* __restrict__ gb, float* __restrict__ gate,
                            int Cn, int Ln)
{
  __shared__ float red[256];
  const int bb = blockIdx.x;
  const int c  = threadIdx.x;
  const float* p = x + (long long)bb * Cn * Ln + (long long)c * Ln;
  float s = 0.f;
  for (int l = 0; l < Ln; ++l) s += p[l];
  red[c] = (s / (float)Ln) * gw[c];
  __syncthreads();
  for (int off = 128; off > 0; off >>= 1) {
    if (c < off) red[c] += red[c + off];
    __syncthreads();
  }
  if (c == 0) gate[bb] = 1.0f / (1.0f + expf(-(red[0] + gb[0])));
}

// Ortho-normalized DFT bases replacing rfft/irfft, generated directly in f16
// (modes interleaved Re,Im):
//  F (L x 2*MMAX):  F[l][2k]=cos/sqrt(L), F[l][2k+1]=-sin/sqrt(L)     (rfft)
//  G (2*MMAX x L):  G[2k][l]=a_k cos/sqrt(L), G[2k+1][l]=-a_k sin/..  (irfft, a_0=1 else 2)
__global__ void basis_kernel(_Float16* __restrict__ Fh, _Float16* __restrict__ Gh,
                             int Ln, int MM)
{
  int idx = blockIdx.x * blockDim.x + threadIdx.x;
  if (idx >= Ln * MM) return;
  int k = idx / Ln, l = idx - k * Ln;
  float invs = rsqrtf((float)Ln);
  int t = (k * l) % Ln;                        // exact angle reduction
  float ang = (float)t * (6.28318530717958647692f / (float)Ln);
  float c = cosf(ang), s = sinf(ang);
  Fh[(long long)l * (2 * MM) + 2 * k]     = (_Float16)( c * invs);
  Fh[(long long)l * (2 * MM) + 2 * k + 1] = (_Float16)(-s * invs);
  float a = (k == 0) ? 1.0f : 2.0f;
  Gh[(long long)(2 * k) * Ln + l]     = (_Float16)( a * c * invs);
  Gh[(long long)(2 * k + 1) * Ln + l] = (_Float16)(-a * s * invs);
}

// ---------------------------------------------------------------------------
// Host-side orchestration
// ---------------------------------------------------------------------------
struct BlockP {
  const float *wr, *wi, *cw, *cb, *lng, *lnb, *w1, *b1, *w2, *b2;
  int m;
};

template <int FLAGS>
static void launch_gemm(hipStream_t s, const float* A, const float* B, float* C,
                        int M, int N, int K,
                        int lda, int ldb, int ldbc, int ldc,
                        long long sA, long long sB, long long sC, int nz,
                        const float* bias, const float* pre, long long sPre,
                        const float* post, long long sPost,
                        const float* cat1 = nullptr, const float* cat2 = nullptr,
                        const float* gates = nullptr, int catRows = 0, int gateStride = 0)
{
  dim3 grid((N + TB_N - 1) / TB_N, (M + TB_M - 1) / TB_M, nz);
  wmma_gemm_kernel<FLAGS><<<grid, 256, 0, s>>>(A, B, C, cat1, cat2, gates,
                                               catRows, gateStride,
                                               M, N, K, lda, ldb, ldbc, ldc, sA, sB, sC,
                                               bias, pre, sPre, post, sPost);
}

// One FNO block, in place on X (B,C,L). W1/W2 scratch, XF/YF spectra, FFNH chunk buffer.
static void fno_block(hipStream_t s, const BlockP& p,
                      float* X, float* W1, float* W2,
                      float* XF, float* YF, float* FFNH,
                      const _Float16* BASFh, const _Float16* BASIh)
{
  const int Cn = CN_, Ln = LN_, Bn = BN_;
  const long long BC = (long long)Bn * Cn;            // 32768
  const long long CL = (long long)Cn * Ln;
  const int m = p.m;

  // 1) forward DFT: XF (2m, B*C) = (X viewed (B*C, L)) * F, stored mode-major
  launch_gemm<GF_TRANS_C | GF_B_HALF>(
      s, X, (const float*)BASFh, XF, (int)BC, 2 * m, Ln, Ln, 2 * MMAX, 1, (int)BC,
      0, 0, 0, 1, nullptr, nullptr, 0, nullptr, 0);

  // 2) complex mode mixing, grid.z = mode; weights read strided from wr/wi[i][o][k]
  //    Yr = Xr*Wr - Xi*Wi ; Yi = Xr*Wi + Xi*Wr
  launch_gemm<GF_B_STRIDED>(
      s, XF,      p.wr, YF,      Bn, Cn, Cn, Cn, Cn * m, m, Cn,
      2 * BC, 1, 2 * BC, m, nullptr, nullptr, 0, nullptr, 0);
  launch_gemm<GF_B_STRIDED | GF_ST_SUB>(
      s, XF + BC, p.wi, YF,      Bn, Cn, Cn, Cn, Cn * m, m, Cn,
      2 * BC, 1, 2 * BC, m, nullptr, nullptr, 0, nullptr, 0);
  launch_gemm<GF_B_STRIDED>(
      s, XF,      p.wi, YF + BC, Bn, Cn, Cn, Cn, Cn * m, m, Cn,
      2 * BC, 1, 2 * BC, m, nullptr, nullptr, 0, nullptr, 0);
  launch_gemm<GF_B_STRIDED | GF_ST_ADD>(
      s, XF + BC, p.wr, YF + BC, Bn, Cn, Cn, Cn, Cn * m, m, Cn,
      2 * BC, 1, 2 * BC, m, nullptr, nullptr, 0, nullptr, 0);

  // 3) inverse DFT: W1 (B*C, L) = YF^T (B*C, 2m) * G
  launch_gemm<GF_TRANS_A | GF_B_HALF>(
      s, YF, (const float*)BASIh, W1, (int)BC, Ln, 2 * m, (int)BC, Ln, 1, Ln,
      0, 0, 0, 1, nullptr, nullptr, 0, nullptr, 0);

  // 4) per-batch 1x1 conv fused: W1 = gelu(cw @ X_b + cb + W1) + X   (residual)
  launch_gemm<GF_BIAS_ROW | GF_PRE_ADD | GF_GELU | GF_POST_ADD>(
      s, p.cw, X, W1, Cn, Ln, Cn, Cn, Ln, 1, Ln,
      0, CL, CL, Bn, p.cb, /*pre=*/W1, CL, /*post=*/X, CL);

  // 5) channel LayerNorm: W2 = LN(W1)
  {
    int total = Bn * Ln;
    ln_channels_kernel<<<(total + 255) / 256, 256, 0, s>>>(W1, W2, p.lng, p.lnb, Bn, Cn, Ln);
  }

  // 6) FFN (chunked over batch): X = w2^T @ gelu(w1^T @ W2 + b1) + b2 + W2
  for (int c0 = 0; c0 < Bn; c0 += FCHUNK) {
    launch_gemm<GF_TRANS_A | GF_BIAS_ROW | GF_GELU>(
        s, p.w1, W2 + c0 * CL, FFNH, 4 * Cn, Ln, Cn, 4 * Cn, Ln, 1, Ln,
        0, CL, 4 * CL, FCHUNK, p.b1, nullptr, 0, nullptr, 0);
    launch_gemm<GF_TRANS_A | GF_BIAS_ROW | GF_POST_ADD>(
        s, p.w2, FFNH, X + c0 * CL, Cn, Ln, 4 * Cn, Cn, Ln, 1, Ln,
        0, 4 * CL, CL, FCHUNK, p.b2, nullptr, 0, W2 + c0 * CL, CL);
  }
}

} // anonymous namespace

extern "C" void kernel_launch(void* const* d_in, const int* in_sizes, int n_in,
                              void* d_out, int out_size, void* d_ws, size_t ws_size,
                              hipStream_t stream) {
  (void)in_sizes; (void)out_size;
  if (n_in < 65) return;

  const int Bn = BN_, Cn = CN_, Ln = LN_;
  const long long CL  = (long long)Cn * Ln;
  const long long BCL = (long long)Bn * CL;           // 16,777,216 floats

  auto F = [&](int i) { return (const float*)d_in[i]; };
  const float* x = F(0);

  // --- workspace layout (floats) ---
  float* W = (float*)d_ws;
  size_t o = 0;
  _Float16* BASFh = (_Float16*)(W + o); o += (size_t)Ln * MMAX;   // f16: L x 2*MMAX
  _Float16* BASIh = (_Float16*)(W + o); o += (size_t)Ln * MMAX;   // f16: 2*MMAX x L
  float* W0    = W + o; o += (size_t)BCL;             // working x
  float* W1    = W + o; o += (size_t)BCL;
  float* W2    = W + o; o += (size_t)BCL;
  float* BR0   = W + o; o += (size_t)BCL;             // logic branch out
  float* BR1   = W + o; o += (size_t)BCL;             // imag branch out
  float* BR2   = W + o; o += (size_t)BCL;             // mem branch out
  float* FFNH  = W + o; o += (size_t)FCHUNK * 4 * Cn * Ln;
  float* XF    = W + o; o += (size_t)2 * MMAX * Bn * Cn;
  float* YF    = W + o; o += (size_t)2 * MMAX * Bn * Cn;
  float* GATES = W + o; o += 3 * (size_t)Bn;
  if (ws_size < o * sizeof(float)) return;            // needs ~450 MiB

  // --- DFT bases (f16, consumed via async-to-LDS staging) ---
  {
    int total = Ln * MMAX;
    basis_kernel<<<(total + 255) / 256, 256, 0, stream>>>(BASFh, BASIh, Ln, MMAX);
  }

  // input ordering: setup_inputs() insertion order
  auto getBlock = [&](int base, int layer, int m) {
    BlockP p;
    p.wr  = F(base + 0) + (size_t)layer * Cn * Cn * m;
    p.wi  = F(base + 1) + (size_t)layer * Cn * Cn * m;
    p.cw  = F(base + 2) + (size_t)layer * Cn * Cn;
    p.cb  = F(base + 3) + (size_t)layer * Cn;
    p.lng = F(base + 4) + (size_t)layer * Cn;
    p.lnb = F(base + 5) + (size_t)layer * Cn;
    p.w1  = F(base + 6) + (size_t)layer * Cn * 4 * Cn;
    p.b1  = F(base + 7) + (size_t)layer * 4 * Cn;
    p.w2  = F(base + 8) + (size_t)layer * 4 * Cn * Cn;
    p.b2  = F(base + 9) + (size_t)layer * Cn;
    p.m = m;
    return p;
  };

  // --- encoder (4 blocks, m=64) ---
  (void)hipMemcpyAsync(W0, x, (size_t)BCL * sizeof(float), hipMemcpyDeviceToDevice, stream);
  for (int i = 0; i < 4; ++i)
    fno_block(stream, getBlock(1, i, 64), W0, W1, W2, XF, YF, FFNH, BASFh, BASIh);

  // --- cognitive branches: logic(m=16), imag(m=32), mem(m=64) ---
  const int   bbase[3]  = {11, 21, 31};
  const int   bmode[3]  = {16, 32, 64};
  float*      bbuf[3]   = {BR0, BR1, BR2};
  const float* gws[3]   = {F(51), F(53), F(55)};
  const float* gbs[3]   = {F(52), F(54), F(56)};
  for (int br = 0; br < 3; ++br) {
    (void)hipMemcpyAsync(bbuf[br], W0, (size_t)BCL * sizeof(float),
                         hipMemcpyDeviceToDevice, stream);
    for (int i = 0; i < 2; ++i)
      fno_block(stream, getBlock(bbase[br], i, bmode[br]),
                bbuf[br], W1, W2, XF, YF, FFNH, BASFh, BASIh);
    gate_kernel<<<Bn, 256, 0, stream>>>(bbuf[br], gws[br], gbs[br],
                                        GATES + br * Bn, Cn, Ln);
  }

  // --- active binding: gate-scaled concat fused into GEMM1 B-loader ---
  const float* bw1 = F(57); const float* bb1 = F(58);
  const float* bw2 = F(59); const float* bb2 = F(60);
  for (int c0 = 0; c0 < Bn; c0 += FCHUNK) {
    launch_gemm<GF_TRANS_A | GF_BIAS_ROW | GF_GELU | GF_B_CAT3>(
        stream, bw1, BR0 + c0 * CL, FFNH, 2 * Cn, Ln, 3 * Cn,
        2 * Cn, Ln, 1, Ln, 0, CL, 2 * CL, FCHUNK,
        bb1, nullptr, 0, nullptr, 0,
        BR1 + c0 * CL, BR2 + c0 * CL, GATES + c0, Cn, Bn);
    launch_gemm<GF_TRANS_A | GF_BIAS_ROW>(
        stream, bw2, FFNH, W1 + c0 * CL, Cn, Ln, 2 * Cn,
        Cn, Ln, 1, Ln, 0, 2 * CL, CL, FCHUNK,
        bb2, nullptr, 0, nullptr, 0);
  }
  {
    int total = Bn * Ln;
    ln_channels_kernel<<<(total + 255) / 256, 256, 0, stream>>>(W1, W0, F(61), F(62),
                                                                Bn, Cn, Ln);
  }

  // --- decoder (2 blocks, m=64) ---
  for (int i = 0; i < 2; ++i)
    fno_block(stream, getBlock(41, i, 64), W0, W1, W2, XF, YF, FFNH, BASFh, BASIh);

  // --- output projection: out (B, L, 512) = W0_b^T @ out_w + out_b ---
  launch_gemm<GF_TRANS_A | GF_BIAS_COL>(
      stream, W0, F(63), (float*)d_out, Ln, 512, Cn,
      Ln, 512, 1, 512, CL, 0, (long long)Ln * 512, Bn,
      F(64), nullptr, 0, nullptr, 0);
}
